// Seq2SeqEncode_39694087750035
// MI455X (gfx1250) — matrix operand order repaired and checked by reading, not compiled, not run on hardware
//
#include <hip/hip_runtime.h>

// ---------------------------------------------------------------------------
// Shapes
// ---------------------------------------------------------------------------
#define VOCAB   32000
#define EMBED   256
#define HIDDEN  512
#define BATCH   32
#define SEQ     512
#define G4H     2048          // 4*HIDDEN
#define MROWS   16384         // BATCH*SEQ
#define NWG     16            // workgroups in the scan kernel

typedef __attribute__((ext_vector_type(16))) __bf16          v16bf;
typedef __attribute__((ext_vector_type(16))) unsigned short  v16su;
typedef __attribute__((ext_vector_type(8)))  unsigned short  v8su;
typedef __attribute__((ext_vector_type(8)))  float           v8f;
typedef __attribute__((ext_vector_type(4)))  float           v4f;

union BF16x16 { v16su u; v16bf b; };

__device__ __forceinline__ unsigned short f2bf(float f) {
    union { float f; unsigned u; } v; v.f = f;
    unsigned r = v.u + 0x7FFFu + ((v.u >> 16) & 1u);   // round-to-nearest-even
    return (unsigned short)(r >> 16);
}
__device__ __forceinline__ float bf2f(unsigned short h) {
    union { unsigned u; float f; } v; v.u = ((unsigned)h) << 16;
    return v.f;
}
__device__ __forceinline__ float sigf(float x) {
    return 1.0f / (1.0f + __expf(-x));
}

// A-fragment packing map (16-bit A 16x32): within a K-chunk of 32,
// offset jj maps to lane-half = jj>>4 and element e = jj&15.
__device__ __forceinline__ void a_frag_pos(int jj, int& lanehalf, int& e) {
    lanehalf = jj >> 4;
    e = jj & 15;
}

// ---------------------------------------------------------------------------
// Repack kernels
// ---------------------------------------------------------------------------
// emb fp32 -> bf16 row-major (one-time; avoids re-converting each row 128x)
__global__ void __launch_bounds__(256)
repack_emb(const float* __restrict__ emb, unsigned short* __restrict__ embbf) {
    int i = blockIdx.x * blockDim.x + threadIdx.x;
    if (i < VOCAB * EMBED) embbf[i] = f2bf(emb[i]);
}

// B packing used here (16-bit B 32x16, same k order as A): lane L holds column N = L%16, 16
// contiguous K values starting at kbase = (L<16 ? 0 : 16).
__global__ void __launch_bounds__(256)
repack_W(const float* __restrict__ W, unsigned short* __restrict__ Wpk) {
    int lf = blockIdx.x * blockDim.x + threadIdx.x;     // 128 nt * 8 ks * 32 lanes
    if (lf >= 128 * 8 * 32) return;
    int lane = lf & 31, ks = (lf >> 5) & 7, nt = lf >> 8;
    int n = nt * 16 + (lane & 15);
    int kbase = ks * 32 + ((lane & 16) ? 16 : 0);
    unsigned short* dst = Wpk + (size_t)lf * 16;
#pragma unroll
    for (int e = 0; e < 16; ++e)
        dst[e] = f2bf(W[(size_t)(kbase + e) * G4H + n]);
}

// U packed per scan-workgroup (contiguous 128KB slice per wg -> LDS memcpy).
__global__ void __launch_bounds__(256)
repack_U(const float* __restrict__ U, unsigned short* __restrict__ Upk) {
    int lf = blockIdx.x * blockDim.x + threadIdx.x;     // 16 wg * 8 ntl * 16 ks * 32
    if (lf >= 65536) return;
    int lane = lf & 31, ks = (lf >> 5) & 15, ntl = (lf >> 9) & 7, wg = lf >> 12;
    int g = ntl >> 1, hh = ntl & 1;
    int n = g * HIDDEN + wg * 32 + hh * 16 + (lane & 15);
    int kbase = ks * 32 + ((lane & 16) ? 16 : 0);
    unsigned short* dst = Upk + (size_t)lf * 16;
#pragma unroll
    for (int e = 0; e < 16; ++e)
        dst[e] = f2bf(U[(size_t)(kbase + e) * G4H + n]);
}

__global__ void __launch_bounds__(256)
repack_Wd(const float* __restrict__ Wd, unsigned short* __restrict__ Wdpk) {
    int lf = blockIdx.x * blockDim.x + threadIdx.x;     // 32 nt * 16 ks * 32 lanes
    if (lf >= 32 * 16 * 32) return;
    int lane = lf & 31, ks = (lf >> 5) & 15, nt = lf >> 9;
    int n = nt * 16 + (lane & 15);
    int kbase = ks * 32 + ((lane & 16) ? 16 : 0);
    unsigned short* dst = Wdpk + (size_t)lf * 16;
#pragma unroll
    for (int e = 0; e < 16; ++e)
        dst[e] = f2bf(Wd[(size_t)(kbase + e) * HIDDEN + n]);
}

// Pack h0 into the bf16 A-fragment double buffer (buf 0) and zero the barrier.
__global__ void __launch_bounds__(256)
init_state(const float* __restrict__ h0, unsigned short* __restrict__ hApk,
           unsigned* __restrict__ counter) {
    int i = blockIdx.x * blockDim.x + threadIdx.x;
    if (i < BATCH * HIDDEN) {
        int bb = i >> 9, k = i & 511;
        int mtl = bb >> 4, r = bb & 15;
        int ks = k >> 5, jj = k & 31;
        int lh, e; a_frag_pos(jj, lh, e);
        int lane = r + (lh << 4);
        hApk[((size_t)(mtl * 16 + ks) * 32 + lane) * 16 + e] = f2bf(h0[i]);
    }
    if (i == 0) *counter = 0u;
}

// ---------------------------------------------------------------------------
// Kernel 1: zx = emb[x] @ W + b, rows ordered m = t*32 + b so the scan reads
// 2 contiguous M-tiles per step. Output stored as packed bf16 D fragments.
// ---------------------------------------------------------------------------
__global__ void __launch_bounds__(256)
embed_proj(const int* __restrict__ x, const unsigned short* __restrict__ embbf,
           const unsigned short* __restrict__ Wpk, const float* __restrict__ bias,
           unsigned short* __restrict__ zxpk) {
    const int wave = threadIdx.x >> 5, lane = threadIdx.x & 31;
    const int tile = blockIdx.x * 8 + wave;     // 1024 mt * 128 nt
    const int mt = tile >> 7, nt = tile & 127;
    const int r = lane & 15;
    const int m = mt * 16 + r;                  // m = t*32 + b
    const int b = m & 31, t = m >> 5;
    int tok = x[b * SEQ + t];
    if (tok < 0) tok = 0;
    if (tok >= VOCAB) tok = VOCAB - 1;
    const unsigned short* erow = embbf + (size_t)tok * EMBED;
    const int n = nt * 16 + r;
    const float bv = bias[n];
    v8f acc;
#pragma unroll
    for (int v = 0; v < 8; ++v) acc[v] = bv;
    const int khalf = (lane & 16) ? 16 : 0;
#pragma unroll
    for (int ks = 0; ks < 8; ++ks) {
        const int k0 = ks * 32 + khalf;
        BF16x16 a, bf;
        v8su lo = *(const v8su*)(erow + k0);
        v8su hi = *(const v8su*)(erow + k0 + 8);
#pragma unroll
        for (int e = 0; e < 8; ++e) { a.u[e] = lo[e]; a.u[e + 8] = hi[e]; }
        bf.u = *(const v16su*)(Wpk + ((size_t)(nt * 8 + ks) * 512 + lane * 16));
        acc = __builtin_amdgcn_wmma_f32_16x16x32_bf16(
            false, a.b, false, bf.b, (short)0, acc, false, false);
    }
    v8su outv;
#pragma unroll
    for (int v = 0; v < 8; ++v) outv[v] = f2bf(acc[v]);
    *(v8su*)(zxpk + ((size_t)(mt * 128 + nt) * 256 + lane * 8)) = outv;
}

// ---------------------------------------------------------------------------
// Kernel 2: sequential LSTM scan. 16 persistent WGs x 256 threads (8 waves).
// h lives in a packed bf16 A-fragment double buffer (hApk, 2 x 32KB): the
// K-loop is pure fragment loads + WMMA, no conversions. WG wg produces the
// ks==wg K-chunk of the next h buffer directly from the gate math.
// ---------------------------------------------------------------------------
__global__ void __launch_bounds__(256)
lstm_scan(const unsigned short* __restrict__ zxpk,
          const unsigned short* __restrict__ Upk,
          const float* __restrict__ c0,
          unsigned short* __restrict__ hApk,     // 2 * 16384 bf16 (packed frags)
          unsigned short* __restrict__ hs,       // [B*T, H] bf16 row-major
          float* __restrict__ dout_tail,         // h_T (16384) then c_T (16384)
          unsigned* __restrict__ counter) {
    extern __shared__ char smem[];
    unsigned short* ldsU = (unsigned short*)smem;              // 131072 B
    float* zbuf = (float*)(smem + 131072);                     // 4*32*32 fp32
    float* cbuf = (float*)(smem + 131072 + 16384);             // 32*32 fp32

    const int tid = threadIdx.x;
    const int wg = blockIdx.x;                 // 0..15
    const int wave = tid >> 5;                 // == local N-tile id
    const int lane = tid & 31;
    const int gate = wave >> 1, hh = wave & 1;
    const int j0 = wg * 32;

    // Preload this WG's U slice into LDS (contiguous 128KB).
    {
        const v8su* src = (const v8su*)(Upk + (size_t)wg * 65536);
        v8su* dst = (v8su*)ldsU;
        for (int i = tid; i < 8192; i += 256) dst[i] = src[i];
    }
#pragma unroll
    for (int e = 0; e < 4; ++e) {              // init cell state
        int lin = e * 256 + tid;               // lin = b*32 + jj
        cbuf[lin] = c0[(lin >> 5) * HIDDEN + j0 + (lin & 31)];
    }
    __syncthreads();

    const int ntg = gate * 32 + wg * 2 + hh;   // global N-tile into zxpk
    const int r = lane & 15;
    const int mrow_add = (lane & 16) ? 8 : 0;
    const int jj_me = hh * 16 + r;

    for (int t = 0; t < SEQ; ++t) {
        // Accumulators initialized from precomputed zx (+bias) fragments.
        v8f acc0, acc1;
        {
            v8su z0 = *(const v8su*)(zxpk + ((size_t)((2 * t + 0) * 128 + ntg) * 256 + lane * 8));
            v8su z1 = *(const v8su*)(zxpk + ((size_t)((2 * t + 1) * 128 + ntg) * 256 + lane * 8));
#pragma unroll
            for (int v = 0; v < 8; ++v) { acc0[v] = bf2f(z0[v]); acc1[v] = bf2f(z1[v]); }
        }
        if (t + 1 < SEQ) {                     // prefetch next step's zx frags
            __builtin_prefetch(zxpk + ((size_t)((2 * t + 2) * 128 + ntg) * 256 + lane * 8), 0, 1);
            __builtin_prefetch(zxpk + ((size_t)((2 * t + 3) * 128 + ntg) * 256 + lane * 8), 0, 1);
        }
        const unsigned short* hA = hApk + (size_t)(t & 1) * 16384;
#pragma unroll 4
        for (int ks = 0; ks < 16; ++ks) {
            BF16x16 a0, a1, bfr;
            a0.u = *(const v16su*)(hA + ((size_t)(ks * 32 + lane)) * 16);         // mtl=0
            a1.u = *(const v16su*)(hA + ((size_t)(512 + ks * 32 + lane)) * 16);   // mtl=1
            bfr.u = *(const v16su*)(ldsU + ((size_t)(wave * 16 + ks) * 512 + lane * 16));
            acc0 = __builtin_amdgcn_wmma_f32_16x16x32_bf16(
                false, a0.b, false, bfr.b, (short)0, acc0, false, false);
            acc1 = __builtin_amdgcn_wmma_f32_16x16x32_bf16(
                false, a1.b, false, bfr.b, (short)0, acc1, false, false);
        }
        // Scatter z tiles to LDS: zbuf[gate*1024 + b*32 + jj]
#pragma unroll
        for (int v = 0; v < 8; ++v) {
            int b0 = v + mrow_add;
            zbuf[gate * 1024 + b0 * 32 + jj_me]        = acc0[v];
            zbuf[gate * 1024 + (b0 + 16) * 32 + jj_me] = acc1[v];
        }
        __syncthreads();

        // Gate nonlinearities + state update (1024 (b,j) elems, 4 per thread).
        unsigned short* hnx = hApk + (size_t)((t + 1) & 1) * 16384;
#pragma unroll
        for (int e = 0; e < 4; ++e) {
            int lin = e * 256 + tid;
            int bb = lin >> 5, jj = lin & 31;
            float zi = zbuf[lin];
            float zf = zbuf[1024 + lin];
            float zg = zbuf[2048 + lin];
            float zo = zbuf[3072 + lin];
            float c = sigf(zf) * cbuf[lin] + sigf(zi) * tanhf(zg);
            float h = sigf(zo) * tanhf(c);
            cbuf[lin] = c;
            unsigned short hb = f2bf(h);
            // packed A-fragment position: this WG supplies K-chunk ks == wg
            int mtl = bb >> 4, rr = bb & 15;
            int lh, ee; a_frag_pos(jj, lh, ee);
            hnx[((size_t)(mtl * 16 + wg) * 32 + (rr + (lh << 4))) * 16 + ee] = hb;
            hs[(size_t)(bb * SEQ + t) * HIDDEN + j0 + jj] = hb;
            if (t == SEQ - 1) {
                dout_tail[bb * HIDDEN + j0 + jj] = h;
                dout_tail[BATCH * HIDDEN + bb * HIDDEN + j0 + jj] = c;
            }
        }
        // Grid barrier (monotonic counter, target = NWG*(t+1)).
        __threadfence();
        __syncthreads();
        if (tid == 0) {
            atomicAdd(counter, 1u);
            const unsigned target = (unsigned)NWG * (unsigned)(t + 1);
            while (__hip_atomic_load(counter, __ATOMIC_ACQUIRE,
                                     __HIP_MEMORY_SCOPE_AGENT) < target) {
                __builtin_amdgcn_s_sleep(2);
            }
        }
        __syncthreads();
    }
}

// ---------------------------------------------------------------------------
// Kernel 3: out = hs @ Wd + bd (fp32 out to d_out)
// ---------------------------------------------------------------------------
__global__ void __launch_bounds__(256)
dense_out(const unsigned short* __restrict__ hs,
          const unsigned short* __restrict__ Wdpk,
          const float* __restrict__ bd,
          float* __restrict__ out) {
    const int wave = threadIdx.x >> 5, lane = threadIdx.x & 31;
    const int tile = blockIdx.x * 8 + wave;     // 1024 mt * 32 nt
    const int mt = tile >> 5, nt = tile & 31;
    const int r = lane & 15;
    const int khalf = (lane & 16) ? 16 : 0;
    const int n = nt * 16 + r;
    const float bv = bd[n];
    v8f acc;
#pragma unroll
    for (int v = 0; v < 8; ++v) acc[v] = bv;
    const unsigned short* arow = hs + (size_t)(mt * 16 + r) * HIDDEN;
#pragma unroll 4
    for (int ks = 0; ks < 16; ++ks) {
        const int k0 = ks * 32 + khalf;
        BF16x16 a, bf;
        v8su lo = *(const v8su*)(arow + k0);
        v8su hi = *(const v8su*)(arow + k0 + 8);
#pragma unroll
        for (int e = 0; e < 8; ++e) { a.u[e] = lo[e]; a.u[e + 8] = hi[e]; }
        bf.u = *(const v16su*)(Wdpk + ((size_t)(nt * 16 + ks) * 512 + lane * 16));
        acc = __builtin_amdgcn_wmma_f32_16x16x32_bf16(
            false, a.b, false, bf.b, (short)0, acc, false, false);
    }
    const int m0 = mt * 16 + ((lane & 16) ? 8 : 0);
#pragma unroll
    for (int v = 0; v < 8; ++v)
        out[(size_t)(m0 + v) * HIDDEN + n] = acc[v];
}

// ---------------------------------------------------------------------------
// Launch
// ---------------------------------------------------------------------------
extern "C" void kernel_launch(void* const* d_in, const int* in_sizes, int n_in,
                              void* d_out, int out_size, void* d_ws, size_t ws_size,
                              hipStream_t stream) {
    (void)in_sizes; (void)n_in; (void)out_size; (void)ws_size;
    const int*   x   = (const int*)  d_in[0];
    const float* h0  = (const float*)d_in[1];
    const float* c0  = (const float*)d_in[2];
    const float* emb = (const float*)d_in[3];
    const float* W   = (const float*)d_in[4];
    const float* U   = (const float*)d_in[5];
    const float* b   = (const float*)d_in[6];
    const float* Wd  = (const float*)d_in[7];
    const float* bd  = (const float*)d_in[8];
    float* out = (float*)d_out;

    // Workspace layout (~100.1 MB total)
    char* ws = (char*)d_ws;
    unsigned short* zxpk  = (unsigned short*)ws;                          // 64 MB
    unsigned short* hs    = (unsigned short*)(ws + ((size_t)64  << 20));  // 16 MB
    unsigned short* embbf = (unsigned short*)(ws + ((size_t)80  << 20));  // 15.7 MB
    unsigned short* Wpk   = (unsigned short*)(ws + ((size_t)96  << 20));  // 1 MB
    unsigned short* Upk   = (unsigned short*)(ws + ((size_t)97  << 20));  // 2 MB
    unsigned short* Wdpk  = (unsigned short*)(ws + ((size_t)99  << 20));  // 0.5 MB
    unsigned short* hApk  = (unsigned short*)(ws + ((size_t)100 << 20));  // 64 KB
    unsigned* counter     = (unsigned*)      (ws + ((size_t)100 << 20) + 65536);

    repack_emb<<<(VOCAB * EMBED + 255) / 256, 256, 0, stream>>>(emb, embbf);
    repack_W <<<(128 * 8 * 32 + 255) / 256, 256, 0, stream>>>(W, Wpk);
    repack_U <<<(65536 + 255) / 256,        256, 0, stream>>>(U, Upk);
    repack_Wd<<<(32 * 16 * 32 + 255) / 256, 256, 0, stream>>>(Wd, Wdpk);
    init_state<<<64, 256, 0, stream>>>(h0, hApk, counter);

    embed_proj<<<16384, 256, 0, stream>>>(x, embbf, Wpk, b, zxpk);

    const size_t smem = 131072 + 16384 + 4096;   // U slice + zbuf + cbuf = 148 KB
    lstm_scan<<<NWG, 256, smem, stream>>>(zxpk, Upk, c0, hApk, hs,
                                          out + (size_t)MROWS * HIDDEN, counter);

    dense_out<<<4096, 256, 0, stream>>>(hs, Wdpk, bd, out);
}
